// QuantizerSelfAttention_35622458753613
// MI455X (gfx1250) — matrix operand order repaired and checked by reading, not compiled, not run on hardware
//
#include <hip/hip_runtime.h>
#include <hip/hip_bf16.h>

typedef __attribute__((ext_vector_type(16))) __bf16 v16bf;
typedef __attribute__((ext_vector_type(8)))  __bf16 v8bf;
typedef __attribute__((ext_vector_type(8)))  float  v8f;

#define P_TOT 4096      // N*H*W
#define QKV_N 768

// LDS weight panel: 64 columns x 256 K, bf16, row stride padded to 264 elem (528 B)
#define BPANEL_STRIDE 264
#define BPANEL_ELEMS  (64 * BPANEL_STRIDE)

union bfrag { v16bf v; v8bf h[2]; };

// ---- pack x: NCHW f32 -> [pixel][C] bf16 ------------------------------------
__global__ void natten_pack_x(const float* __restrict__ x, __bf16* __restrict__ xp) {
    int idx = blockIdx.x * blockDim.x + threadIdx.x;     // 4096*256 threads
    int p = idx >> 8, c = idx & 255;
    int n = p >> 10, hw = p & 1023;
    xp[idx] = (__bf16)x[((n * 256 + c) << 10) + hw];
}

// ---- generic f32 -> bf16 convert -------------------------------------------
__global__ void natten_cvt_bf16(const float* __restrict__ src, __bf16* __restrict__ dst, int count) {
    int i = blockIdx.x * blockDim.x + threadIdx.x;
    if (i < count) dst[i] = (__bf16)src[i];
}

// ---- stage 64x256 weight panel into LDS via the gfx1250 async-LDS engine ----
__device__ __forceinline__ void stage_weight_panel(const __bf16* __restrict__ W,
                                                   int tn0, __bf16* shB) {
    const char* wbytes = (const char*)W;
    unsigned lbase = (unsigned)(size_t)(void*)shB;   // low 32 bits of generic ptr = LDS offset
#pragma unroll
    for (int it = 0; it < 8; ++it) {
        int chunk = it * 256 + threadIdx.x;          // 2048 x 16B chunks = 32 KB
        int col = chunk >> 5;                        // 32 chunks per 512B source row
        int ko  = (chunk & 31) << 4;                 // byte offset within row
        const char* g = wbytes + (size_t)(tn0 + col) * 512 + ko;
        unsigned laddr = lbase + col * 528 + ko;     // padded LDS row stride
        asm volatile("global_load_async_to_lds_b128 %0, %1, off"
                     :: "v"(laddr), "v"(g) : "memory");
    }
    asm volatile("s_wait_asynccnt 0x0" ::: "memory");
    __syncthreads();
}

// ---- 16x64-per-wave WMMA accumulation over K=256 ----------------------------
// All four B fragments are loaded into distinct registers before the WMMAs so
// the scheduler can front-load the 8 ds_load_b128s and stagger s_wait_dscnt,
// instead of gating each v_wmma on the immediately preceding LDS pair.
__device__ __forceinline__ void wmma_16x64(const __bf16* __restrict__ ap,
                                           const __bf16* shB,
                                           int row, int hi, v8f acc[4]) {
    const __bf16* bbase = shB + row * BPANEL_STRIDE + hi * 16;
#pragma unroll
    for (int kk = 0; kk < 256; kk += 32) {
        bfrag a;
        a.h[0] = *(const v8bf*)(ap + kk + hi * 16);        // A K-chunks hi*16, +8
        a.h[1] = *(const v8bf*)(ap + kk + hi * 16 + 8);
        bfrag b0, b1, b2, b3;
        const __bf16* bp = bbase + kk;
        b0.h[0] = *(const v8bf*)(bp);
        b0.h[1] = *(const v8bf*)(bp + 8);
        b1.h[0] = *(const v8bf*)(bp + 16 * BPANEL_STRIDE);
        b1.h[1] = *(const v8bf*)(bp + 16 * BPANEL_STRIDE + 8);
        b2.h[0] = *(const v8bf*)(bp + 32 * BPANEL_STRIDE);
        b2.h[1] = *(const v8bf*)(bp + 32 * BPANEL_STRIDE + 8);
        b3.h[0] = *(const v8bf*)(bp + 48 * BPANEL_STRIDE);
        b3.h[1] = *(const v8bf*)(bp + 48 * BPANEL_STRIDE + 8);
        acc[0] = __builtin_amdgcn_wmma_f32_16x16x32_bf16(
            false, a.v, false, b0.v, (short)0, acc[0], false, false);
        acc[1] = __builtin_amdgcn_wmma_f32_16x16x32_bf16(
            false, a.v, false, b1.v, (short)0, acc[1], false, false);
        acc[2] = __builtin_amdgcn_wmma_f32_16x16x32_bf16(
            false, a.v, false, b2.v, (short)0, acc[2], false, false);
        acc[3] = __builtin_amdgcn_wmma_f32_16x16x32_bf16(
            false, a.v, false, b3.v, (short)0, acc[3], false, false);
    }
}

// ---- QKV GEMM: [4096x256] @ [768x256]^T -> f32 [4096x768] (+bias, q*0.125) --
// block = 8 waves = 128x64 tile; weights staged in LDS via async-LDS engine
__global__ void natten_qkv_gemm(const __bf16* __restrict__ A,   // xp [4096][256]
                                const __bf16* __restrict__ W,   // wq [768][256]
                                const float*  __restrict__ bias,
                                float* __restrict__ out) {      // qkv [4096][768]
    __shared__ __align__(16) __bf16 shB[BPANEL_ELEMS];
    int tn0 = blockIdx.y * 64;
    stage_weight_panel(W, tn0, shB);

    int wv = threadIdx.x >> 5, lane = threadIdx.x & 31;
    int row = lane & 15, hi = lane >> 4;
    int m0w = blockIdx.x * 128 + wv * 16;
    const __bf16* ap = A + (m0w + row) * 256;

    v8f acc[4] = {};
    wmma_16x64(ap, shB, row, hi, acc);

    int m0 = m0w + hi * 8;
#pragma unroll
    for (int t = 0; t < 4; ++t) {
        int col = tn0 + t * 16 + row;
        float bv = bias[col];
        float scale = (col < 256) ? 0.125f : 1.0f;   // q /= sqrt(e=64)
#pragma unroll
        for (int r = 0; r < 8; ++r)
            out[(m0 + r) * QKV_N + col] = (acc[t][r] + bv) * scale;
    }
}

// ---- attention core: one thread per (pixel, head), online softmax -----------
__global__ void natten_attn(const float* __restrict__ qkv,   // [4096][768]
                            const float* __restrict__ rpb,   // [4][13][13]
                            __bf16* __restrict__ op) {       // [4096][256]
    int t = blockIdx.x * blockDim.x + threadIdx.x;   // 16384
    int p = t >> 2, head = t & 3;
    int n = p >> 10, hw = p & 1023;
    int i = hw >> 5, j = hw & 31;
    int si = i - 3; si = si < 0 ? 0 : (si > 25 ? 25 : si);
    int sj = j - 3; sj = sj < 0 ? 0 : (sj > 25 ? 25 : sj);

    const float* qp = qkv + p * QKV_N + head * 64;
    float q[64];
#pragma unroll
    for (int e = 0; e < 64; ++e) q[e] = qp[e];

    const float* rp = rpb + head * 169;
    float mval = -1e30f, l = 0.f;
    float o[64];
#pragma unroll
    for (int e = 0; e < 64; ++e) o[e] = 0.f;

    for (int a = 0; a < 7; ++a) {
        int r = si + a;
        int rbase = (r - i + 6) * 13;
        for (int b = 0; b < 7; ++b) {
            int c2 = sj + b;
            int p2 = ((n << 5) + r) * 32 + c2;
            const float* kp = qkv + p2 * QKV_N + 256 + head * 64;
            float s = rp[rbase + (c2 - j + 6)];
#pragma unroll
            for (int e = 0; e < 64; ++e) s += q[e] * kp[e];
            float mnew = fmaxf(mval, s);
            float corr = __expf(mval - mnew);
            float w    = __expf(s - mnew);
            l = l * corr + w;
            const float* vp = kp + 256;
#pragma unroll
            for (int e = 0; e < 64; ++e) o[e] = o[e] * corr + w * vp[e];
            mval = mnew;
        }
    }
    float inv = 1.0f / l;
    __bf16* od = op + p * 256 + head * 64;
#pragma unroll
    for (int e = 0; e < 64; ++e) od[e] = (__bf16)(o[e] * inv);
}

// ---- out GEMM: [4096x256] @ [256x256]^T + bias + residual, NCHW scatter -----
__global__ void natten_out_gemm(const __bf16* __restrict__ A,   // op [4096][256]
                                const __bf16* __restrict__ W,   // wo [256][256]
                                const float*  __restrict__ bias,
                                const float*  __restrict__ x,   // residual, NCHW
                                float* __restrict__ out) {      // NCHW
    __shared__ __align__(16) __bf16 shB[BPANEL_ELEMS];
    int tn0 = blockIdx.y * 64;
    stage_weight_panel(W, tn0, shB);

    int wv = threadIdx.x >> 5, lane = threadIdx.x & 31;
    int row = lane & 15, hi = lane >> 4;
    int m0w = blockIdx.x * 128 + wv * 16;
    const __bf16* ap = A + (m0w + row) * 256;

    v8f acc[4] = {};
    wmma_16x64(ap, shB, row, hi, acc);

    int m0 = m0w + hi * 8;
#pragma unroll
    for (int t = 0; t < 4; ++t) {
        int col = tn0 + t * 16 + row;
        float bv = bias[col];
#pragma unroll
        for (int r = 0; r < 8; ++r) {
            int pidx = m0 + r;
            int n = pidx >> 10, hwp = pidx & 1023;
            int oi = ((n * 256 + col) << 10) + hwp;
            out[oi] = acc[t][r] + bv + x[oi];
        }
    }
}

extern "C" void kernel_launch(void* const* d_in, const int* in_sizes, int n_in,
                              void* d_out, int out_size, void* d_ws, size_t ws_size,
                              hipStream_t stream) {
    const float* x     = (const float*)d_in[0];   // (4,256,32,32)
    const float* qkv_w = (const float*)d_in[1];   // (768,256)
    const float* qkv_b = (const float*)d_in[2];   // (768,)
    const float* rpb   = (const float*)d_in[3];   // (4,13,13)
    const float* out_w = (const float*)d_in[4];   // (256,256)
    const float* out_b = (const float*)d_in[5];   // (256,)
    float* out = (float*)d_out;

    char* ws = (char*)d_ws;
    __bf16* xp   = (__bf16*)(ws);                      // 4096*256*2  = 2,097,152
    __bf16* wq   = (__bf16*)(ws + 2097152);            //  768*256*2  =   393,216
    __bf16* wo   = (__bf16*)(ws + 2490368);            //  256*256*2  =   131,072
    float*  qkv  = (float* )(ws + 2621440);            // 4096*768*4  = 12,582,912
    __bf16* op   = (__bf16*)(ws + 15204352);           // 4096*256*2  = 2,097,152

    natten_pack_x  <<<4096, 256, 0, stream>>>(x, xp);
    natten_cvt_bf16<<< 768, 256, 0, stream>>>(qkv_w, wq, 768 * 256);
    natten_cvt_bf16<<< 256, 256, 0, stream>>>(out_w, wo, 256 * 256);
    // block = 128 rows x 64 cols; grid = (4096/128, 768/64)
    natten_qkv_gemm<<<dim3(32, 12), 256, 0, stream>>>(xp, wq, qkv_b, qkv);
    // 16384 (pixel,head) threads
    natten_attn    <<<  64, 256, 0, stream>>>(qkv, rpb, op);
    // grid = (4096/128, 256/64)
    natten_out_gemm<<<dim3(32, 4), 256, 0, stream>>>(op, wo, out_b, x, out);
}
